// EfficientAttention_68599217651965
// MI455X (gfx1250) — matrix operand order repaired and checked by reading, not compiled, not run on hardware
//
#include <hip/hip_runtime.h>

// Problem constants (fixed by the reference)
constexpr int BATCH  = 2;
constexpr int CC     = 256;   // channels
constexpr int RED    = 32;    // reduced channels
constexpr int NPTS   = 6272;  // 8*28*28
constexpr int ITILES = NPTS / 16;   // 392 query tiles per batch
constexpr int NCHUNK = NPTS / 32;   // 196 j-chunks of 32
constexpr int JWAVES = 4;           // waves cooperating on one i-tile
constexpr float LOG2E = 1.4426950408889634f;

typedef __attribute__((ext_vector_type(16))) __bf16 v16bf;
typedef __attribute__((ext_vector_type(8)))  __bf16 v8bf;
typedef __attribute__((ext_vector_type(2)))  __bf16 v2bf;
typedef __attribute__((ext_vector_type(8)))  float  v8f;
typedef __attribute__((ext_vector_type(2)))  float  v2f;
typedef __attribute__((ext_vector_type(8)))  unsigned v8u;

__device__ inline float fast_exp2(float v) { return __builtin_amdgcn_exp2f(v); }

// ---------------------------------------------------------------------------
// Kernel 1: 1x1x1 conv projections.  q = qw@x (scaled by 1/sqrt(32)), k = kw@x,
// v = vw@x.  Q,K stored [B][N][32] bf16 (row i contiguous over r) so they load
// directly as WMMA A (K rows) / B (Q cols) operands; V stored [B][32][N] bf16
// (row r contiguous over j) so it loads directly as WMMA A tiles for O=V*P^T.
// Weight addresses are wave-uniform -> scalar loads from constant cache.
// ---------------------------------------------------------------------------
__global__ __launch_bounds__(256) void ea_qkv_kernel(
    const float* __restrict__ x,
    const float* __restrict__ qw, const float* __restrict__ kw,
    const float* __restrict__ vw,
    __bf16* __restrict__ Q, __bf16* __restrict__ K, __bf16* __restrict__ V)
{
    int gid = blockIdx.x * 256 + threadIdx.x;   // one thread per (b, i)
    int b = gid / NPTS;
    int i = gid - b * NPTS;
    const float* xp = x + (size_t)b * CC * NPTS + i;

    float acc[RED];
    const float scale = 0.17677669529663687f;   // 1/sqrt(32)

    // ---- Q ----
    #pragma unroll
    for (int r = 0; r < RED; ++r) acc[r] = 0.f;
    for (int c = 0; c < CC; ++c) {
        float xv = xp[(size_t)c * NPTS];
        #pragma unroll
        for (int r = 0; r < RED; ++r) acc[r] += qw[r * CC + c] * xv;
    }
    #pragma unroll
    for (int r = 0; r < RED; ++r)
        Q[((size_t)b * NPTS + i) * RED + r] = (__bf16)(acc[r] * scale);

    // ---- K ----
    #pragma unroll
    for (int r = 0; r < RED; ++r) acc[r] = 0.f;
    for (int c = 0; c < CC; ++c) {
        float xv = xp[(size_t)c * NPTS];
        #pragma unroll
        for (int r = 0; r < RED; ++r) acc[r] += kw[r * CC + c] * xv;
    }
    #pragma unroll
    for (int r = 0; r < RED; ++r)
        K[((size_t)b * NPTS + i) * RED + r] = (__bf16)acc[r];

    // ---- V (transposed store) ----
    #pragma unroll
    for (int r = 0; r < RED; ++r) acc[r] = 0.f;
    for (int c = 0; c < CC; ++c) {
        float xv = xp[(size_t)c * NPTS];
        #pragma unroll
        for (int r = 0; r < RED; ++r) acc[r] += vw[r * CC + c] * xv;
    }
    #pragma unroll
    for (int r = 0; r < RED; ++r)
        V[((size_t)b * RED + r) * NPTS + i] = (__bf16)acc[r];
}

// Load one WMMA 16-bit A-matrix row (16xK=32) slice for this lane from a
// row-major (K contiguous) row pointer.  Layout: lanes 0-15 hold
// K={0..15}, lanes 16-31 hold K={16..31}.
__device__ inline v16bf load_a32(const __bf16* row, int half)
{
    v8bf lo = *(const v8bf*)(row + half * 16);
    v8bf hi = *(const v8bf*)(row + half * 16 + 8);
    return __builtin_shufflevector(lo, hi, 0,1,2,3,4,5,6,7,8,9,10,11,12,13,14,15);
}

// Pair conversion expressed so clang can emit a single v_cvt_pk_bf16_f32.
__device__ inline unsigned pack_bf16x2(float lo, float hi)
{
    v2f f = {lo, hi};
    v2bf c = __builtin_convertvector(f, v2bf);
    return __builtin_bit_cast(unsigned, c);
}

// ---------------------------------------------------------------------------
// Kernel 2: split-j flash attention + output projection + residual.
// 4 waves cooperate on each 16-query tile (each streams an interleaved quarter
// of the j-chunks), then merge online-softmax partials (m, l, O) through LDS.
// S^T = K*Q^T via WMMA so softmax-over-j is lane-local + one shfl_xor(16);
// P^T C-layout -> B-layout conversion exchanges packed bf16x2 dwords.
// The softmax denominator is computed as an all-ones WMMA column-sum of P^T
// (moves a 15-op VALU reduction onto the matrix pipe and covers both halves).
// Block = 256 threads = 8 waves = 2 tiles.
// ---------------------------------------------------------------------------
__global__ __launch_bounds__(256) void ea_attn_kernel(
    const __bf16* __restrict__ Q, const __bf16* __restrict__ K,
    const __bf16* __restrict__ V, const float* __restrict__ ow,
    const float* __restrict__ x, float* __restrict__ out)
{
    __shared__ float s_o[2][16][JWAVES][32];
    __shared__ float s_m[2][JWAVES][32];
    __shared__ float s_l[2][JWAVES][32];

    int wid  = threadIdx.x >> 5;        // 0..7
    int t    = wid >> 2;                // tile slot within block (0/1)
    int w    = wid & (JWAVES - 1);      // j-split wave index (0..3)
    int lane = threadIdx.x & 31;
    int half = lane >> 4;
    int l16  = lane & 15;

    int tile = blockIdx.x * 2 + t;      // 0..783
    int b    = tile / ITILES;
    int i0   = (tile - b * ITILES) * 16;

    const __bf16* Qb = Q + (size_t)b * NPTS * RED;
    const __bf16* Kb = K + (size_t)b * NPTS * RED;
    const __bf16* Vb = V + (size_t)b * RED * NPTS;

    // Q tile as B-matrix (32 r x 16 i): lane col i = i0+l16, K=r contiguous.
    v16bf qb = *(const v16bf*)(Qb + (size_t)(i0 + l16) * RED + half * 16);

    // All-ones bf16 A-tile for the column-sum WMMA.
    v8u onesbits = {0x3f803f80u, 0x3f803f80u, 0x3f803f80u, 0x3f803f80u,
                    0x3f803f80u, 0x3f803f80u, 0x3f803f80u, 0x3f803f80u};
    v16bf ones16 = __builtin_bit_cast(v16bf, onesbits);

    v8f o0 = {0,0,0,0,0,0,0,0};     // O^T rows r=0..15  (col i = l16)
    v8f o1 = {0,0,0,0,0,0,0,0};     // O^T rows r=16..31
    float m_run = -3.0e38f, l_run = 0.f;

    #pragma clang loop unroll_count(2)
    for (int cidx = w; cidx < NCHUNK; cidx += JWAVES) {
        int j0 = cidx * 32;
        // K^T A-tiles: rows j (16 each), K-dim = r = 32.
        v16bf ka0 = load_a32(Kb + (size_t)(j0 + l16) * RED, half);
        v16bf ka1 = load_a32(Kb + (size_t)(j0 + 16 + l16) * RED, half);
        v8f zero = {0,0,0,0,0,0,0,0};
        v8f s0 = __builtin_amdgcn_wmma_f32_16x16x32_bf16(false, ka0, false, qb,
                                                         (short)0, zero, false, false);
        v8f s1 = __builtin_amdgcn_wmma_f32_16x16x32_bf16(false, ka1, false, qb,
                                                         (short)0, zero, false, false);

        // Shared chunk max per query column (i = l16) across both halves.
        float cmax = s0[0];
        #pragma unroll
        for (int h = 1; h < 8; ++h) cmax = fmaxf(cmax, s0[h]);
        #pragma unroll
        for (int h = 0; h < 8; ++h) cmax = fmaxf(cmax, s1[h]);
        cmax = fmaxf(cmax, __shfl_xor(cmax, 16, 32));
        float m_new = fmaxf(m_run, cmax);

        // exp(s - m) == exp2(fma(s, log2e, -m*log2e)) -> 1 VALU + 1 TRANS/elt.
        float nm2  = -m_new * LOG2E;
        float corr = fast_exp2(__fmaf_rn(m_run, LOG2E, nm2));
        float p0[8], p1[8];
        #pragma unroll
        for (int h = 0; h < 8; ++h) p0[h] = fast_exp2(__fmaf_rn(s0[h], LOG2E, nm2));
        #pragma unroll
        for (int h = 0; h < 8; ++h) p1[h] = fast_exp2(__fmaf_rn(s1[h], LOG2E, nm2));

        // Pack P into bf16x2 dwords, then one exchange per dword (8 total).
        v8u pw;
        #pragma unroll
        for (int hh = 0; hh < 4; ++hh) {
            unsigned a0 = pack_bf16x2(p0[2*hh], p0[2*hh+1]);
            unsigned a1 = pack_bf16x2(p1[2*hh], p1[2*hh+1]);
            unsigned sh0 = __shfl_xor(a0, 16, 32);
            unsigned sh1 = __shfl_xor(a1, 16, 32);
            pw[hh]     = half ? sh1 : a0;   // K slots 0..7
            pw[hh + 4] = half ? a1 : sh0;   // K slots 8..15
        }
        v16bf pb = __builtin_bit_cast(v16bf, pw);

        // Denominator: column-sum of P^T on the matrix pipe.  Every row of D
        // equals the sum over all 32 j, so every lane reads it from D[0].
        v8f lw = __builtin_amdgcn_wmma_f32_16x16x32_bf16(false, ones16, false, pb,
                                                         (short)0, zero, false, false);
        l_run = l_run * corr + lw[0];
        m_run = m_new;

        #pragma unroll
        for (int h = 0; h < 8; ++h) { o0[h] *= corr; o1[h] *= corr; }

        // V A-tiles (rows r, K-dim = j chunk of 32), row r contiguous over j.
        v16bf va0 = load_a32(Vb + (size_t)l16 * NPTS + j0, half);
        v16bf va1 = load_a32(Vb + (size_t)(l16 + 16) * NPTS + j0, half);
        o0 = __builtin_amdgcn_wmma_f32_16x16x32_bf16(false, va0, false, pb,
                                                     (short)0, o0, false, false);
        o1 = __builtin_amdgcn_wmma_f32_16x16x32_bf16(false, va1, false, pb,
                                                     (short)0, o1, false, false);
    }

    // ---- merge the 4 partial results through LDS ----
    s_m[t][w][lane] = m_run;
    s_l[t][w][lane] = l_run;
    #pragma unroll
    for (int h = 0; h < 8; ++h) {
        s_o[t][h][w][lane]     = o0[h];
        s_o[t][h + 8][w][lane] = o1[h];
    }
    __syncthreads();

    float mw[JWAVES];
    float mstar = -3.0e38f;
    #pragma unroll
    for (int u = 0; u < JWAVES; ++u) {
        mw[u] = s_m[t][u][lane];
        mstar = fmaxf(mstar, mw[u]);
    }
    float alpha[JWAVES], lsum = 0.f;
    float nms2 = -mstar * LOG2E;
    #pragma unroll
    for (int u = 0; u < JWAVES; ++u) {
        alpha[u] = fast_exp2(__fmaf_rn(mw[u], LOG2E, nms2));
        lsum += alpha[u] * s_l[t][u][lane];
    }
    float linv = 1.0f / lsum;
    #pragma unroll
    for (int h = 0; h < 8; ++h) {
        float acc0 = 0.f, acc1 = 0.f;
        #pragma unroll
        for (int u = 0; u < JWAVES; ++u) {
            acc0 += alpha[u] * s_o[t][h][u][lane];
            acc1 += alpha[u] * s_o[t][h + 8][u][lane];
        }
        o0[h] = acc0 * linv;
        o1[h] = acc1 * linv;
    }

    // O^T as B-matrix (32 r x 16 i): packed-dword exchange (identical in all
    // 4 waves, which then split the 16 channel tiles of the projection).
    v8u obw;
    #pragma unroll
    for (int hh = 0; hh < 4; ++hh) {
        unsigned a0 = pack_bf16x2(o0[2*hh], o0[2*hh+1]);
        unsigned a1 = pack_bf16x2(o1[2*hh], o1[2*hh+1]);
        unsigned sh0 = __shfl_xor(a0, 16, 32);
        unsigned sh1 = __shfl_xor(a1, 16, 32);
        obw[hh]     = half ? sh1 : a0;
        obw[hh + 4] = half ? a1 : sh0;
    }
    v16bf ob = __builtin_bit_cast(v16bf, obw);

    // y = out_w (256x32) * O^T (32x16) + residual x; wave w does 4 of the 16
    // channel tiles.
    const float* xb = x   + (size_t)b * CC * NPTS;
    float* yb       = out + (size_t)b * CC * NPTS;
    for (int ct = w * 4; ct < w * 4 + 4; ++ct) {
        const float* wr = ow + (size_t)(ct * 16 + l16) * RED;
        v8f wlo = *(const v8f*)(wr + half * 16);
        v8f whi = *(const v8f*)(wr + half * 16 + 8);
        v8bf clo = __builtin_convertvector(wlo, v8bf);
        v8bf chi = __builtin_convertvector(whi, v8bf);
        v16bf wa = __builtin_shufflevector(clo, chi,
                                           0,1,2,3,4,5,6,7,8,9,10,11,12,13,14,15);
        v8f zero = {0,0,0,0,0,0,0,0};
        v8f y = __builtin_amdgcn_wmma_f32_16x16x32_bf16(false, wa, false, ob,
                                                        (short)0, zero, false, false);
        #pragma unroll
        for (int h = 0; h < 8; ++h) {
            int c = ct * 16 + h + 8 * half;           // C-layout row
            size_t idx = (size_t)c * NPTS + i0 + l16; // col i = l16
            yb[idx] = xb[idx] + y[h];
        }
    }
}

// ---------------------------------------------------------------------------
extern "C" void kernel_launch(void* const* d_in, const int* in_sizes, int n_in,
                              void* d_out, int out_size, void* d_ws, size_t ws_size,
                              hipStream_t stream)
{
    const float* x  = (const float*)d_in[0];
    const float* qw = (const float*)d_in[1];
    const float* kw = (const float*)d_in[2];
    const float* vw = (const float*)d_in[3];
    const float* ow = (const float*)d_in[4];
    float* out = (float*)d_out;

    __bf16* Q = (__bf16*)d_ws;                       // [B][N][32]
    __bf16* K = Q + (size_t)BATCH * NPTS * RED;      // [B][N][32]
    __bf16* V = K + (size_t)BATCH * NPTS * RED;      // [B][32][N]

    // 12544 (b,i) points, one thread each.
    ea_qkv_kernel<<<(BATCH * NPTS) / 256, 256, 0, stream>>>(x, qw, kw, vw, Q, K, V);

    // 784 query tiles, 4 waves each, 8 waves (2 tiles) per block.
    ea_attn_kernel<<<(BATCH * ITILES) / 2, 256, 0, stream>>>(Q, K, V, ow, x, out);
}
